// VDCNN_22960895164476
// MI455X (gfx1250) — compile-verified, matrix operand order checked by reading, not run on hardware
//
#include <hip/hip_runtime.h>
#include <hip/hip_bf16.h>
#include <math.h>

typedef __attribute__((ext_vector_type(16))) _Float16 v16h;
typedef __attribute__((ext_vector_type(8)))  _Float16 v8h;
typedef __attribute__((ext_vector_type(8)))  float    v8f;

#define SEGS 120

static inline int cdiv(int a, int b) { return (a + b - 1) / b; }

// ---------------------------------------------------------------------------
// Block-wide sum reduction (blockDim.x must be power of two, sh[blockDim.x]).
// ---------------------------------------------------------------------------
__device__ __forceinline__ float block_reduce_sum(float v, float* sh) {
  int t = threadIdx.x;
  sh[t] = v;
  __syncthreads();
  for (int s = blockDim.x >> 1; s > 0; s >>= 1) {
    if (t < s) sh[t] += sh[t + s];
    __syncthreads();
  }
  v = sh[0];
  __syncthreads();
  return v;
}

// ---------------------------------------------------------------------------
// Segment extraction: x[8,1,100,7500] -> seg[120,100,500]
// seg[b*15+si][h][w] = x[b,0,h, si*500 + w]
// ---------------------------------------------------------------------------
__global__ void seg_extract_kernel(const float* __restrict__ x,
                                   float* __restrict__ seg) {
  int idx = blockIdx.x * blockDim.x + threadIdx.x;
  const int total = SEGS * 100 * 500;
  if (idx >= total) return;
  int w = idx % 500;
  int h = (idx / 500) % 100;
  int s = idx / (500 * 100);
  int b = s / 15, si = s % 15;
  seg[idx] = x[((size_t)b * 100 + h) * 7500 + (size_t)si * 500 + w];
}

// ---------------------------------------------------------------------------
// Implicit-GEMM 5x5 "same" conv via WMMA f16 -> f32, double-buffered LDS.
//   per segment: D[COUT, H*W] = W[COUT, CIN*25] x im2col[CIN*25, H*W] + bias
// Block = 128 threads (4 waves). Block tile: 64 pixels x COUT.
// Each wave owns a 16-pixel N tile and loops COUT/16 M tiles.
// LDS panels are laid out so every fragment is two aligned 16-byte chunks
// (2x ds_load_b128 per operand, no scalar gathers).
// ---------------------------------------------------------------------------
template <int CIN, int COUT, int H, int W>
__global__ __launch_bounds__(128) void conv2d_stem_wmma(
    const float* __restrict__ in,    // [SEGS][CIN][H][W]
    const float* __restrict__ wgt,   // [COUT][CIN*25]
    const float* __restrict__ bias,  // [COUT]
    float* __restrict__ out) {       // [SEGS][COUT][H*W]
  constexpr int K = CIN * 25;
  constexpr int P = H * W;
  constexpr int MT = COUT / 16;
  constexpr int KSTEPS = (K + 31) / 32;

  // A panel: [buf][m][k]            rows of 64 B (16B-aligned chunks)
  // B panel: [buf][pixel][k] (+pad) rows of 80 B (16B-aligned chunks)
  __shared__ _Float16 ldsA[2][COUT][32];
  __shared__ _Float16 ldsB[2][64][40];

  const int seg = blockIdx.y;
  const int p0 = blockIdx.x * 64;
  const int tid = threadIdx.x;
  const int lane = tid & 31;
  const int wid = tid >> 5;

  const float* inS = in + (size_t)seg * CIN * P;

  auto stageA = [&](int ks, int buf) {
    const int k0 = ks * 32;
    for (int e = tid; e < COUT * 32; e += 128) {
      int m = e >> 5, kk = e & 31, k = k0 + kk;
      float v = (k < K) ? wgt[(size_t)m * K + k] : 0.f;
      ldsA[buf][m][kk] = (_Float16)v;
    }
  };
  auto stageB = [&](int ks, int buf) {
    const int k0 = ks * 32;
    for (int e = tid; e < 32 * 64; e += 128) {
      int pl = e & 63, kk = e >> 6, k = k0 + kk;  // pixel fastest: coalesced
      float v = 0.f;
      int pix = p0 + pl;
      if (k < K && pix < P) {
        int oy = pix / W, ox = pix % W;
        int ci = k / 25, rs = k % 25;
        int iy = oy + rs / 5 - 2, ix = ox + rs % 5 - 2;
        if (iy >= 0 && iy < H && ix >= 0 && ix < W)
          v = inS[((size_t)ci * H + iy) * W + ix];
      }
      ldsB[buf][pl][kk] = (_Float16)v;  // transposed: frag reads contiguous
    }
  };

  // prologue: stage K-step 0 into buffer 0
  stageA(0, 0);
  stageB(0, 0);
  __syncthreads();

  v8f acc[MT] = {};

  for (int ks = 0; ks < KSTEPS; ++ks) {
    const int cur = ks & 1;
    // software pipeline: stage next K-step into the alternate buffer
    if (ks + 1 < KSTEPS) {
      stageA(ks + 1, cur ^ 1);
      stageB(ks + 1, cur ^ 1);
    }
    // prefetch input lines for K-step ks+2 toward the WGP caches
    if (ks + 2 < KSTEPS && tid < 32) {
      int k = (ks + 2) * 32 + tid;
      if (k < K) __builtin_prefetch(&inS[(size_t)(k / 25) * P], 0, 1);
    }

    // ---- fragment loads: two aligned v8h LDS chunks per operand ----
    const int lo = lane & 15;
    const int khi = (lane >> 4) * 8;  // lanes 16-31 hold K offset +8 / +24
    v8h b0 = *(const v8h*)&ldsB[cur][wid * 16 + lo][khi];
    v8h b1 = *(const v8h*)&ldsB[cur][wid * 16 + lo][16 + khi];
    v16h bfr = __builtin_shufflevector(b0, b1, 0, 1, 2, 3, 4, 5, 6, 7, 8, 9,
                                       10, 11, 12, 13, 14, 15);
#pragma unroll
    for (int mt = 0; mt < MT; ++mt) {
      v8h a0 = *(const v8h*)&ldsA[cur][mt * 16 + lo][khi];
      v8h a1 = *(const v8h*)&ldsA[cur][mt * 16 + lo][16 + khi];
      v16h afr = __builtin_shufflevector(a0, a1, 0, 1, 2, 3, 4, 5, 6, 7, 8, 9,
                                         10, 11, 12, 13, 14, 15);
      acc[mt] = __builtin_amdgcn_wmma_f32_16x16x32_f16(
          /*neg_a=*/false, afr, /*neg_b=*/false, bfr,
          /*c_mod=*/(short)0, acc[mt], /*reuse_a=*/false, /*reuse_b=*/false);
    }
    __syncthreads();
  }

  // ---- epilogue: D layout -> N = lane&15, M = 8*(lane>>4) + vgpr ----
  const int lo = lane & 15;
  const int mrow = (lane >> 4) * 8;
  const int pix = p0 + wid * 16 + lo;
  if (pix < P) {
#pragma unroll
    for (int mt = 0; mt < MT; ++mt) {
#pragma unroll
      for (int r = 0; r < 8; ++r) {
        int co = mt * 16 + mrow + r;
        out[((size_t)seg * COUT + co) * P + pix] = acc[mt][r] + bias[co];
      }
    }
  }
}

// ---------------------------------------------------------------------------
// Per-(segment,channel) mean/var over P spatial positions (train-mode BN2d).
// grid.x = SEGS*C, block = 256.
// ---------------------------------------------------------------------------
__global__ __launch_bounds__(256) void bn2d_stats_kernel(
    const float* __restrict__ x, float* __restrict__ mean,
    float* __restrict__ var, int P) {
  __shared__ float sh[256];
  const int sc = blockIdx.x;
  const float* p = x + (size_t)sc * P;
  float s = 0.f, s2 = 0.f;
  for (int i = threadIdx.x; i < P; i += 256) {
    float v = p[i];
    s += v;
    s2 += v * v;
  }
  s = block_reduce_sum(s, sh);
  s2 = block_reduce_sum(s2, sh);
  if (threadIdx.x == 0) {
    float m = s / P;
    mean[sc] = m;
    var[sc] = s2 / P - m * m;
  }
}

// ---------------------------------------------------------------------------
// Fused BN2d(apply) + ReLU + 2x2/2 maxpool.
// ---------------------------------------------------------------------------
template <int C, int H, int W>
__global__ void bn_relu_pool2d_kernel(const float* __restrict__ x,
                                      const float* __restrict__ mean,
                                      const float* __restrict__ var,
                                      const float* __restrict__ g,
                                      const float* __restrict__ be,
                                      float* __restrict__ y) {
  constexpr int HO = H / 2, WO = W / 2;
  int idx = blockIdx.x * blockDim.x + threadIdx.x;
  const int total = SEGS * C * HO * WO;
  if (idx >= total) return;
  int wo = idx % WO;
  int t = idx / WO;
  int ho = t % HO;
  t /= HO;
  int c = t % C;
  int seg = t / C;
  int sc = seg * C + c;
  float m = mean[sc];
  float rs = rsqrtf(var[sc] + 1e-5f);
  float gg = g[c] * rs;
  float bb = be[c] - gg * m;
  const float* base = x + ((size_t)sc * H + ho * 2) * W + wo * 2;
  float a = gg * base[0] + bb;
  float b2 = gg * base[1] + bb;
  float cc = gg * base[W] + bb;
  float dd = gg * base[W + 1] + bb;
  float mx = fmaxf(fmaxf(a, b2), fmaxf(cc, dd));
  y[idx] = fmaxf(mx, 0.f);
}

// ---------------------------------------------------------------------------
// fc5: feat[seg][j] = dot(pool4[seg][:], w[j][:]) + b[j], then write into
// res[b][32][15] layout: res[(seg/15)*32 + j][seg%15].
// grid = (120, 32), block = 256.
// ---------------------------------------------------------------------------
__global__ __launch_bounds__(256) void fc5_kernel(const float* __restrict__ in,
                                                  const float* __restrict__ w,
                                                  const float* __restrict__ b,
                                                  float* __restrict__ res) {
  __shared__ float sh[256];
  const int seg = blockIdx.x, j = blockIdx.y;
  const int KK = 32 * 6 * 31;  // 5952
  const float* a = in + (size_t)seg * KK;
  const float* ww = w + (size_t)j * KK;
  float s = 0.f;
  for (int i = threadIdx.x; i < KK; i += 256) s += a[i] * ww[i];
  s = block_reduce_sum(s, sh);
  if (threadIdx.x == 0) {
    int bb = seg / 15, si = seg % 15;
    res[((size_t)bb * 32 + j) * 15 + si] = s + b[j];
  }
}

// ---------------------------------------------------------------------------
// conv1d, kernel=3, pad=1.  grid.x = B*Co, block = 32 (thread per position).
// ---------------------------------------------------------------------------
__global__ void conv1d_k3_kernel(const float* __restrict__ in,
                                 const float* __restrict__ w,
                                 const float* __restrict__ b,
                                 float* __restrict__ out, int B, int Ci, int Co,
                                 int L) {
  int bc = blockIdx.x;
  int bb = bc / Co, co = bc % Co;
  int l = threadIdx.x;
  if (l >= L) return;
  float s = b[co];
  for (int ci = 0; ci < Ci; ++ci) {
    const float* ip = in + ((size_t)bb * Ci + ci) * L;
    const float* wp = w + ((size_t)co * Ci + ci) * 3;
    float x0 = (l > 0) ? ip[l - 1] : 0.f;
    float x1 = ip[l];
    float x2 = (l < L - 1) ? ip[l + 1] : 0.f;
    s += wp[0] * x0 + wp[1] * x1 + wp[2] * x2;
  }
  out[((size_t)bb * Co + co) * L + l] = s;
}

// ---------------------------------------------------------------------------
// Train-mode BN1d over (B,L) per channel + ReLU, in place.
// grid.x = C, block = 128.
// ---------------------------------------------------------------------------
__global__ __launch_bounds__(128) void bn1d_relu_kernel(
    float* __restrict__ x, const float* __restrict__ g,
    const float* __restrict__ be, int B, int C, int L) {
  __shared__ float sh[128];
  const int c = blockIdx.x;
  const int n = B * L;
  float s = 0.f, s2 = 0.f;
  for (int i = threadIdx.x; i < n; i += 128) {
    int b = i / L, l = i % L;
    float v = x[((size_t)b * C + c) * L + l];
    s += v;
    s2 += v * v;
  }
  s = block_reduce_sum(s, sh);
  s2 = block_reduce_sum(s2, sh);
  float m = s / n;
  float rs = rsqrtf(fmaxf(s2 / n - m * m, 0.f) + 1e-5f);
  float gg = g[c] * rs;
  float bb = be[c] - gg * m;
  for (int i = threadIdx.x; i < n; i += 128) {
    int b = i / L, l = i % L;
    size_t o = ((size_t)b * C + c) * L + l;
    x[o] = fmaxf(gg * x[o] + bb, 0.f);
  }
}

// ---------------------------------------------------------------------------
// maxpool1d kernel=3 stride=2 pad=1.
// ---------------------------------------------------------------------------
__global__ void maxpool1d_kernel(const float* __restrict__ x,
                                 float* __restrict__ y, int B, int C, int L,
                                 int LO) {
  int idx = blockIdx.x * blockDim.x + threadIdx.x;
  int total = B * C * LO;
  if (idx >= total) return;
  int lo = idx % LO;
  int t = idx / LO;
  int c = t % C;
  int b = t / C;
  const float* p = x + ((size_t)b * C + c) * L;
  int s = lo * 2 - 1;
  float m = -3.4e38f;
  for (int i = 0; i < 3; ++i) {
    int j = s + i;
    if (j >= 0 && j < L) m = fmaxf(m, p[j]);
  }
  y[idx] = m;
}

// ---------------------------------------------------------------------------
// Adaptive max pool 1d: L -> O.
// ---------------------------------------------------------------------------
__global__ void adamax1d_kernel(const float* __restrict__ x,
                                float* __restrict__ y, int B, int C, int L,
                                int O) {
  int idx = blockIdx.x * blockDim.x + threadIdx.x;
  int total = B * C * O;
  if (idx >= total) return;
  int o = idx % O;
  int t = idx / O;
  int c = t % C;
  int b = t / C;
  int s = (o * L) / O;
  int e = ((o + 1) * L + O - 1) / O;
  const float* p = x + ((size_t)b * C + c) * L;
  float m = p[s];
  for (int i = s + 1; i < e; ++i) m = fmaxf(m, p[i]);
  y[idx] = m;
}

// ---------------------------------------------------------------------------
// Dense layer: out[n][m] = (relu?) dot(in[n], w[m]) + b[m].
// ---------------------------------------------------------------------------
__global__ void fc_kernel(const float* __restrict__ in,
                          const float* __restrict__ w,
                          const float* __restrict__ b, float* __restrict__ out,
                          int N, int M, int K, int relu) {
  int idx = blockIdx.x * blockDim.x + threadIdx.x;
  if (idx >= N * M) return;
  int n = idx / M, m = idx % M;
  const float* a = in + (size_t)n * K;
  const float* ww = w + (size_t)m * K;
  float s = b[m];
  for (int k = 0; k < K; ++k) s += a[k] * ww[k];
  out[idx] = relu ? fmaxf(s, 0.f) : s;
}

// ---------------------------------------------------------------------------
// Host orchestration.
// ---------------------------------------------------------------------------
extern "C" void kernel_launch(void* const* d_in, const int* in_sizes, int n_in,
                              void* d_out, int out_size, void* d_ws,
                              size_t ws_size, hipStream_t stream) {
  (void)in_sizes; (void)n_in; (void)out_size; (void)ws_size;
  // ---- inputs (setup_inputs dict order) ----
  const float* x = (const float*)d_in[0];
  const float* s_w[4]  = {(const float*)d_in[1], (const float*)d_in[5],
                          (const float*)d_in[9], (const float*)d_in[13]};
  const float* s_b[4]  = {(const float*)d_in[2], (const float*)d_in[6],
                          (const float*)d_in[10], (const float*)d_in[14]};
  const float* s_g[4]  = {(const float*)d_in[3], (const float*)d_in[7],
                          (const float*)d_in[11], (const float*)d_in[15]};
  const float* s_be[4] = {(const float*)d_in[4], (const float*)d_in[8],
                          (const float*)d_in[12], (const float*)d_in[16]};
  const float* fc5_w = (const float*)d_in[17];
  const float* fc5_b = (const float*)d_in[18];
  const float* c0_w = (const float*)d_in[19];
  const float* c0_b = (const float*)d_in[20];
  // blocks[i]: w1,b1,g1,be1,w2,b2,g2,be2 at 21 + 8*i
  const float* bw1[4]; const float* bb1[4]; const float* bg1[4]; const float* bbe1[4];
  const float* bw2[4]; const float* bb2[4]; const float* bg2[4]; const float* bbe2[4];
  for (int i = 0; i < 4; ++i) {
    int base = 21 + 8 * i;
    bw1[i] = (const float*)d_in[base + 0];
    bb1[i] = (const float*)d_in[base + 1];
    bg1[i] = (const float*)d_in[base + 2];
    bbe1[i] = (const float*)d_in[base + 3];
    bw2[i] = (const float*)d_in[base + 4];
    bb2[i] = (const float*)d_in[base + 5];
    bg2[i] = (const float*)d_in[base + 6];
    bbe2[i] = (const float*)d_in[base + 7];
  }
  const float* fc1_w = (const float*)d_in[53];
  const float* fc1_b = (const float*)d_in[54];
  const float* fc2_w = (const float*)d_in[55];
  const float* fc2_b = (const float*)d_in[56];
  const float* fc3_w = (const float*)d_in[57];
  const float* fc3_b = (const float*)d_in[58];

  // ---- workspace carve ----
  char* ws = (char*)d_ws;
  size_t off = 0;
  auto carve = [&](size_t floats) -> float* {
    float* p = (float*)(ws + off);
    off = (off + floats * sizeof(float) + 255) & ~(size_t)255;
    return p;
  };
  float* seg0 = carve((size_t)SEGS * 100 * 500);        // 6.0M
  float* conv = carve((size_t)SEGS * 16 * 100 * 500);   // 96M (reused all layers)
  float* poolA = carve((size_t)SEGS * 16 * 50 * 250);   // 24M (ping)
  float* poolB = carve((size_t)SEGS * 32 * 25 * 125);   // 12M (pong)
  float* meanb = carve((size_t)SEGS * 64);
  float* varb = carve((size_t)SEGS * 64);
  float* res = carve(8 * 32 * 15);
  float* tA = carve(8 * 512 * 16);
  float* tB = carve(8 * 512 * 16);

  // ---- segment extraction ----
  seg_extract_kernel<<<cdiv(SEGS * 100 * 500, 256), 256, 0, stream>>>(x, seg0);

  // ---- 2D stem: conv(WMMA) -> BN stats -> BN+ReLU+pool ----
  // L1: 1 -> 16, 100x500
  conv2d_stem_wmma<1, 16, 100, 500>
      <<<dim3(cdiv(100 * 500, 64), SEGS), 128, 0, stream>>>(seg0, s_w[0], s_b[0], conv);
  bn2d_stats_kernel<<<SEGS * 16, 256, 0, stream>>>(conv, meanb, varb, 100 * 500);
  bn_relu_pool2d_kernel<16, 100, 500>
      <<<cdiv(SEGS * 16 * 50 * 250, 256), 256, 0, stream>>>(conv, meanb, varb,
                                                            s_g[0], s_be[0], poolA);
  // L2: 16 -> 32, 50x250
  conv2d_stem_wmma<16, 32, 50, 250>
      <<<dim3(cdiv(50 * 250, 64), SEGS), 128, 0, stream>>>(poolA, s_w[1], s_b[1], conv);
  bn2d_stats_kernel<<<SEGS * 32, 256, 0, stream>>>(conv, meanb, varb, 50 * 250);
  bn_relu_pool2d_kernel<32, 50, 250>
      <<<cdiv(SEGS * 32 * 25 * 125, 256), 256, 0, stream>>>(conv, meanb, varb,
                                                            s_g[1], s_be[1], poolB);
  // L3: 32 -> 64, 25x125
  conv2d_stem_wmma<32, 64, 25, 125>
      <<<dim3(cdiv(25 * 125, 64), SEGS), 128, 0, stream>>>(poolB, s_w[2], s_b[2], conv);
  bn2d_stats_kernel<<<SEGS * 64, 256, 0, stream>>>(conv, meanb, varb, 25 * 125);
  bn_relu_pool2d_kernel<64, 25, 125>
      <<<cdiv(SEGS * 64 * 12 * 62, 256), 256, 0, stream>>>(conv, meanb, varb,
                                                           s_g[2], s_be[2], poolA);
  // L4: 64 -> 32, 12x62
  conv2d_stem_wmma<64, 32, 12, 62>
      <<<dim3(cdiv(12 * 62, 64), SEGS), 128, 0, stream>>>(poolA, s_w[3], s_b[3], conv);
  bn2d_stats_kernel<<<SEGS * 32, 256, 0, stream>>>(conv, meanb, varb, 12 * 62);
  bn_relu_pool2d_kernel<32, 12, 62>
      <<<cdiv(SEGS * 32 * 6 * 31, 256), 256, 0, stream>>>(conv, meanb, varb,
                                                          s_g[3], s_be[3], poolB);

  // ---- fc5 + transpose into res[8][32][15] ----
  fc5_kernel<<<dim3(SEGS, 32), 256, 0, stream>>>(poolB, fc5_w, fc5_b, res);

  // ---- 1D tower ----
  // c0: 32 -> 64, L=15
  conv1d_k3_kernel<<<8 * 64, 32, 0, stream>>>(res, c0_w, c0_b, tA, 8, 32, 64, 15);
  // block0 (64->64, L=15)
  conv1d_k3_kernel<<<8 * 64, 32, 0, stream>>>(tA, bw1[0], bb1[0], tB, 8, 64, 64, 15);
  bn1d_relu_kernel<<<64, 128, 0, stream>>>(tB, bg1[0], bbe1[0], 8, 64, 15);
  conv1d_k3_kernel<<<8 * 64, 32, 0, stream>>>(tB, bw2[0], bb2[0], tA, 8, 64, 64, 15);
  bn1d_relu_kernel<<<64, 128, 0, stream>>>(tA, bg2[0], bbe2[0], 8, 64, 15);
  maxpool1d_kernel<<<cdiv(8 * 64 * 8, 256), 256, 0, stream>>>(tA, tB, 8, 64, 15, 8);
  // block1 (64->128, L=8)
  conv1d_k3_kernel<<<8 * 128, 32, 0, stream>>>(tB, bw1[1], bb1[1], tA, 8, 64, 128, 8);
  bn1d_relu_kernel<<<128, 128, 0, stream>>>(tA, bg1[1], bbe1[1], 8, 128, 8);
  conv1d_k3_kernel<<<8 * 128, 32, 0, stream>>>(tA, bw2[1], bb2[1], tB, 8, 128, 128, 8);
  bn1d_relu_kernel<<<128, 128, 0, stream>>>(tB, bg2[1], bbe2[1], 8, 128, 8);
  maxpool1d_kernel<<<cdiv(8 * 128 * 4, 256), 256, 0, stream>>>(tB, tA, 8, 128, 8, 4);
  // block2 (128->256, L=4)
  conv1d_k3_kernel<<<8 * 256, 32, 0, stream>>>(tA, bw1[2], bb1[2], tB, 8, 128, 256, 4);
  bn1d_relu_kernel<<<256, 128, 0, stream>>>(tB, bg1[2], bbe1[2], 8, 256, 4);
  conv1d_k3_kernel<<<8 * 256, 32, 0, stream>>>(tB, bw2[2], bb2[2], tA, 8, 256, 256, 4);
  bn1d_relu_kernel<<<256, 128, 0, stream>>>(tA, bg2[2], bbe2[2], 8, 256, 4);
  maxpool1d_kernel<<<cdiv(8 * 256 * 2, 256), 256, 0, stream>>>(tA, tB, 8, 256, 4, 2);
  // block3 (256->512, L=2)
  conv1d_k3_kernel<<<8 * 512, 32, 0, stream>>>(tB, bw1[3], bb1[3], tA, 8, 256, 512, 2);
  bn1d_relu_kernel<<<512, 128, 0, stream>>>(tA, bg1[3], bbe1[3], 8, 512, 2);
  conv1d_k3_kernel<<<8 * 512, 32, 0, stream>>>(tA, bw2[3], bb2[3], tB, 8, 512, 512, 2);
  bn1d_relu_kernel<<<512, 128, 0, stream>>>(tB, bg2[3], bbe2[3], 8, 512, 2);
  // adaptive max 2 -> 8 : [8,512,8] -> flatten [8,4096]
  adamax1d_kernel<<<cdiv(8 * 512 * 8, 256), 256, 0, stream>>>(tB, tA, 8, 512, 2, 8);

  // ---- head ----
  fc_kernel<<<cdiv(8 * 2048, 256), 256, 0, stream>>>(tA, fc1_w, fc1_b, tB, 8, 2048, 4096, 1);
  fc_kernel<<<cdiv(8 * 2048, 256), 256, 0, stream>>>(tB, fc2_w, fc2_b, tA, 8, 2048, 2048, 1);
  fc_kernel<<<1, 256, 0, stream>>>(tA, fc3_w, fc3_b, (float*)d_out, 8, 2, 2048, 0);
}
